// SeparateAttention_76493367541979
// MI455X (gfx1250) — matrix operand order repaired and checked by reading, not compiled, not run on hardware
//
#include <hip/hip_runtime.h>
#include <math.h>
#include <stdint.h>

// ---- problem constants -----------------------------------------------------
#define B_ 4
#define T_ 24
#define N_ 325
#define E_ 128
#define H_ 8
#define D_ 16
#define M_ (B_*T_*N_)          /* 31200 rows */
#define MTILES (M_/16)         /* 1950, exact */
#define SCALE 0.25f            /* D^-0.5 */
#define WH_ELEMS (E_*E_)       /* 16384 f16 = 32 KB, pre-swizzled weights */

typedef __attribute__((ext_vector_type(16))) _Float16 v16h;
typedef __attribute__((ext_vector_type(8)))  float    v8f;
typedef unsigned int v4u __attribute__((ext_vector_type(4)));
typedef int          v8i __attribute__((ext_vector_type(8)));
typedef int          v4i __attribute__((ext_vector_type(4)));

#if defined(__has_builtin)
#  if __has_builtin(__builtin_amdgcn_tensor_load_to_lds) && \
      __has_builtin(__builtin_amdgcn_s_wait_tensorcnt)
#    define HAVE_TDM 1
#  endif
#endif
#ifndef HAVE_TDM
#  define HAVE_TDM 0
#endif

// ---------------------------------------------------------------------------
// Convert W (fp32, [f,e] row-major) to f16, pre-swizzled into the exact
// per-lane WMMA B-operand order:  element(kk,nt,lane,i) = W[nt*16 + (lane&15)]
// [kk*32 + (lane>>4)*16 + i].  The resulting 32 KB block is a byte-for-byte
// image of what each GEMM block wants in LDS -> one contiguous TDM copy.
// ---------------------------------------------------------------------------
__global__ __launch_bounds__(256)
void convert_w(const float* __restrict__ W, _Float16* __restrict__ Wh)
{
    const int idx = blockIdx.x * 256 + threadIdx.x;   // 0 .. 16383
    const int i   = idx & 15;
    const int l   = (idx >> 4) & 31;
    const int nt  = (idx >> 9) & 7;
    const int kk  = idx >> 12;
    const int row = nt * 16 + (l & 15);
    const int col = kk * 32 + (l >> 4) * 16 + i;
    Wh[idx] = (_Float16)W[row * E_ + col];
}

// ---------------------------------------------------------------------------
// GEMM: Y[m,f] = relu?( sum_e (X[m,e]+X2[m,e]) * W[f,e] + bias[f] ), K=N=128.
// Pre-swizzled f16 W staged once per block into LDS (TDM tensor_load_to_lds),
// shared by 8 waves; each wave computes a 16(M) x 128(N) strip with
// v_wmma_f32_16x16x32_f16 (fp32 accumulate).
// ---------------------------------------------------------------------------
__global__ __launch_bounds__(256)
void gemm128_wmma(const float* __restrict__ X, const float* __restrict__ X2,
                  const _Float16* __restrict__ Wh, const float* __restrict__ bias,
                  float* __restrict__ Y, int relu)
{
    __shared__ __align__(32) _Float16 ldsW[WH_ELEMS];   // 32 KB

    const int lane = threadIdx.x & 31;
    const int wave = threadIdx.x >> 5;

#if HAVE_TDM
    if (wave == 0) {
        // Build D# (ISA ch.8): group0 = {count=1 | lds_addr | global_addr | type=2}
        const uint32_t laddr = (uint32_t)(uintptr_t)(&ldsW[0]);
        const uint64_t gaddr = (uint64_t)(uintptr_t)Wh;
        v4u g0 = {0u, 0u, 0u, 0u};
        g0[0] = 1u;                                          // count=1, no gather
        g0[1] = laddr;                                       // lds_addr [63:32]
        g0[2] = (uint32_t)gaddr;                             // global_addr lo
        g0[3] = (uint32_t)((gaddr >> 32) & 0x01FFFFFFull)    // global_addr [56:32]
              | (2u << 30);                                  // type=2 ("image")
        // group1: data_size=4B, tensor_dim0=tile_dim0=8192 dwords (1-row tile),
        // tensor_dim1=tile_dim1=1, dim0 stride = 8192.
        v8i g1 = {0, 0, 0, 0, 0, 0, 0, 0};
        g1[0] = 0x00020000;     // workgroup_mask=0 | data_size=2 (4B)
        g1[1] = 0x20000000;     // tensor_dim0.lo16 = 8192  (bits 63:48)
        g1[2] = 0x00010000;     // tensor_dim0.hi16=0 | tensor_dim1.lo16=1
        g1[3] = 0x20000000;     // tensor_dim1.hi16=0 | tile_dim0=8192
        g1[4] = 0x00000001;     // tile_dim1=1 | tile_dim2=0
        g1[5] = 8192;           // tensor_dim0_stride lo32
        g1[6] = 0x20000000;     // stride0.hi16=0 | tensor_dim1_stride.lo16=8192
        g1[7] = 0;
        v4i z4 = {0, 0, 0, 0};              // groups 2/3 unused (<=2D tensor)
        v8i z8 = {0, 0, 0, 0, 0, 0, 0, 0};  // unused trailing group (6-arg form)
        __builtin_amdgcn_tensor_load_to_lds(g0, g1, z4, z4, z8, 0);
    }
    __builtin_amdgcn_s_wait_tensorcnt(0);   // waves that issued nothing pass thru
    __syncthreads();
#else
    {
        const float4* src = (const float4*)Wh;   // 2048 x 16B = 32 KB
        float4*       dst = (float4*)ldsW;
        for (int i = threadIdx.x; i < 2048; i += 256) dst[i] = src[i];
        __syncthreads();
    }
#endif

    const int mtile = blockIdx.x * 8 + wave;
    if (mtile >= MTILES) return;              // wave-uniform, after the barrier

    const int lane16 = lane & 15;
    const int half   = lane >> 4;             // 0 or 1

    const size_t rowoff = (size_t)(mtile * 16 + lane16) * E_;
    const float* xrow   = X + rowoff;
    const float* x2row  = X2 ? (X2 + rowoff) : nullptr;
    const v16h*  wl     = (const v16h*)ldsW;  // [kk][nt][lane] -> 16 halves

    v8f acc[8] = {};

    #pragma unroll
    for (int kk = 0; kk < 4; kk++) {
        const int k0 = kk * 32;
        // A tile 16x32 f16: lanes 0-15 -> K k0+0..15,
        // lanes 16-31 -> K k0+16..31 (matches pre-swizzled B).
        v16h a;
        const int kb = half * 16;
        if (x2row) {
            #pragma unroll
            for (int i = 0; i < 8; i++) {
                a[i]     = (_Float16)(xrow[k0 + kb + i]      + x2row[k0 + kb + i]);
                a[i + 8] = (_Float16)(xrow[k0 + 8 + kb + i] + x2row[k0 + 8 + kb + i]);
            }
        } else {
            #pragma unroll
            for (int i = 0; i < 8; i++) {
                a[i]     = (_Float16)xrow[k0 + kb + i];
                a[i + 8] = (_Float16)xrow[k0 + 8 + kb + i];
            }
        }
        #pragma unroll
        for (int nt = 0; nt < 8; nt++) {
            const v16h bm = wl[(kk * 8 + nt) * 32 + lane];
            acc[nt] = __builtin_amdgcn_wmma_f32_16x16x32_f16(
                false, a, false, bm, (short)0, acc[nt], false, false);
        }
    }

    // Epilogue: C/D layout = VGPR j -> row j (lanes 0-15) / row j+8 (lanes 16-31)
    #pragma unroll
    for (int nt = 0; nt < 8; nt++) {
        const int col = nt * 16 + lane16;
        const float bc = bias[col];
        #pragma unroll
        for (int j = 0; j < 8; j++) {
            float vv = acc[nt][j] + bc;
            if (relu) vv = fmaxf(vv, 0.f);
            Y[(size_t)(mtile * 16 + half * 8 + j) * E_ + col] = vv;
        }
    }
}

// ---------------------------------------------------------------------------
// Temporal attention: one wave per (b,n,h).  K/V (24x16) in LDS, each of the
// 24 active lanes owns one query row t; softmax over s in registers.
// ---------------------------------------------------------------------------
__global__ __launch_bounds__(32)
void temporal_attn(const float* __restrict__ q, const float* __restrict__ k,
                   const float* __restrict__ v, float* __restrict__ temporal,
                   float* __restrict__ attn_t)
{
    __shared__ float Ks[T_][D_];
    __shared__ float Vs[T_][D_];
    const int h = blockIdx.x % H_;
    const int n = (blockIdx.x / H_) % N_;
    const int b = blockIdx.x / (H_ * N_);

    for (int i = threadIdx.x; i < T_ * D_; i += 32) {
        const int s = i >> 4, d = i & 15;
        const size_t off = (((size_t)(b * T_ + s) * N_ + n) * E_) + h * D_ + d;
        Ks[s][d] = k[off];
        Vs[s][d] = v[off];
    }
    __syncthreads();

    const int t = threadIdx.x;
    if (t < T_) {
        const size_t qoff = (((size_t)(b * T_ + t) * N_ + n) * E_) + h * D_;
        float qv[D_];
        #pragma unroll
        for (int d = 0; d < D_; d++) qv[d] = q[qoff + d];

        float sc[T_];
        float mx = -3.0e38f;
        for (int s = 0; s < T_; s++) {
            float dot = 0.f;
            #pragma unroll
            for (int d = 0; d < D_; d++) dot += qv[d] * Ks[s][d];
            dot *= SCALE;
            sc[s] = dot;
            mx = fmaxf(mx, dot);
        }
        float l = 0.f;
        for (int s = 0; s < T_; s++) { float e = __expf(sc[s] - mx); sc[s] = e; l += e; }
        const float inv = 1.f / l;

        float o[D_] = {};
        float* arow = attn_t + (((size_t)(b * N_ + n) * H_ + h) * T_ + t) * T_;
        for (int s = 0; s < T_; s++) {
            const float a = sc[s] * inv;
            arow[s] = a;
            #pragma unroll
            for (int d = 0; d < D_; d++) o[d] += a * Vs[s][d];
        }
        #pragma unroll
        for (int d = 0; d < D_; d++) temporal[qoff + d] = o[d];
    }
}

// ---------------------------------------------------------------------------
// Spatial attention: one 256-thread block per (b,t,h).  K/V (325x16 each,
// 41.6 KB) staged in LDS.  Online softmax -> attn_s touched only 2W + 1R.
// ---------------------------------------------------------------------------
__global__ __launch_bounds__(256)
void spatial_attn(const float* __restrict__ q, const float* __restrict__ k,
                  const float* __restrict__ v, const float* __restrict__ adj,
                  float* __restrict__ spatial, float* __restrict__ attn_s)
{
    __shared__ float Ks[N_ * D_];
    __shared__ float Vs[N_ * D_];
    const int h = blockIdx.x % H_;
    const int t = (blockIdx.x / H_) % T_;
    const int b = blockIdx.x / (H_ * T_);

    for (int i = threadIdx.x; i < N_ * D_; i += 256) {
        const int m = i >> 4, d = i & 15;
        const size_t off = (((size_t)(b * T_ + t) * N_ + m) * E_) + h * D_ + d;
        Ks[i] = k[off];
        Vs[i] = v[off];
    }
    __syncthreads();

    for (int n = threadIdx.x; n < N_; n += 256) {
        const size_t qoff = (((size_t)(b * T_ + t) * N_ + n) * E_) + h * D_;
        float qv[D_];
        #pragma unroll
        for (int d = 0; d < D_; d++) qv[d] = q[qoff + d];

        const float* adjrow = adj + (size_t)n * N_;            // adj[0][n][:]
        float* arow = attn_s + (((size_t)(b * T_ + t) * H_ + h) * N_ + n) * (size_t)N_;

        float mx = -3.0e38f, l = 0.f;
        float o[D_] = {};
        for (int m = 0; m < N_; m++) {
            float dot = 0.f;
            #pragma unroll
            for (int d = 0; d < D_; d++) dot += qv[d] * Ks[m * D_ + d];
            const float a = adjrow[m];
            const float s = (a != 0.f) ? dot * SCALE * a : -1.0e9f;
            arow[m] = s;                                       // raw gated score
            if (s > mx) {                                      // online softmax
                const float c = __expf(mx - s);
                l *= c;
                #pragma unroll
                for (int d = 0; d < D_; d++) o[d] *= c;
                mx = s;
            }
            const float e = __expf(s - mx);
            l += e;
            #pragma unroll
            for (int d = 0; d < D_; d++) o[d] += e * Vs[m * D_ + d];
        }
        const float inv = 1.f / l;
        #pragma unroll
        for (int d = 0; d < D_; d++) spatial[qoff + d] = o[d] * inv;
        // normalize the stored row in place
        for (int m = 0; m < N_; m++) arow[m] = __expf(arow[m] - mx) * inv;
    }
}

// ---------------------------------------------------------------------------
extern "C" void kernel_launch(void* const* d_in, const int* in_sizes, int n_in,
                              void* d_out, int out_size, void* d_ws, size_t ws_size,
                              hipStream_t stream)
{
    (void)in_sizes; (void)n_in; (void)out_size; (void)ws_size;
    const float* query = (const float*)d_in[0];
    const float* key_i = (const float*)d_in[1];
    const float* value = (const float*)d_in[2];
    const float* adj   = (const float*)d_in[3];
    const float* Wq = (const float*)d_in[4];  const float* bq = (const float*)d_in[5];
    const float* Wk = (const float*)d_in[6];  const float* bk = (const float*)d_in[7];
    const float* Wv = (const float*)d_in[8];  const float* bv = (const float*)d_in[9];
    const float* Wt = (const float*)d_in[10]; const float* bt = (const float*)d_in[11];
    const float* Wsp= (const float*)d_in[12]; const float* bsp= (const float*)d_in[13];
    const float* Wo = (const float*)d_in[14]; const float* bo = (const float*)d_in[15];

    const size_t SZ = (size_t)M_ * E_;          // 3,993,600 floats
    float* ws  = (float*)d_ws;
    float* qb  = ws;                            // q        (later: t_out)
    float* kb  = ws + SZ;                       // k        (later: s_out)
    float* vb  = ws + 2 * SZ;                   // v
    float* tb  = ws + 3 * SZ;                   // temporal
    float* sb  = ws + 4 * SZ;                   // spatial
    _Float16* wh = (_Float16*)(ws + 5 * SZ);    // 6 x 32 KB pre-swizzled f16 W
    _Float16* WqH = wh + 0 * WH_ELEMS;
    _Float16* WkH = wh + 1 * WH_ELEMS;
    _Float16* WvH = wh + 2 * WH_ELEMS;
    _Float16* WtH = wh + 3 * WH_ELEMS;
    _Float16* WsH = wh + 4 * WH_ELEMS;
    _Float16* WoH = wh + 5 * WH_ELEMS;

    float* out    = (float*)d_out;              // [B,T,N,E]
    float* attn_t = out + SZ;                   // [B*N,H,T,T]
    float* attn_s = attn_t + (size_t)B_ * N_ * H_ * T_ * T_;   // [B*T,H,N,N]

    convert_w<<<WH_ELEMS / 256, 256, 0, stream>>>(Wq,  WqH);
    convert_w<<<WH_ELEMS / 256, 256, 0, stream>>>(Wk,  WkH);
    convert_w<<<WH_ELEMS / 256, 256, 0, stream>>>(Wv,  WvH);
    convert_w<<<WH_ELEMS / 256, 256, 0, stream>>>(Wt,  WtH);
    convert_w<<<WH_ELEMS / 256, 256, 0, stream>>>(Wsp, WsH);
    convert_w<<<WH_ELEMS / 256, 256, 0, stream>>>(Wo,  WoH);

    const int gemmGrid = (MTILES + 7) / 8;      // 8 waves (M-tiles) per block

    gemm128_wmma<<<gemmGrid, 256, 0, stream>>>(query, nullptr, WqH, bq, qb, 0);
    gemm128_wmma<<<gemmGrid, 256, 0, stream>>>(key_i, nullptr, WkH, bk, kb, 0);
    gemm128_wmma<<<gemmGrid, 256, 0, stream>>>(value, nullptr, WvH, bv, vb, 0);

    temporal_attn<<<B_ * N_ * H_, 32, 0, stream>>>(qb, kb, vb, tb, attn_t);
    spatial_attn <<<B_ * T_ * H_, 256, 0, stream>>>(qb, kb, vb, adj, sb, attn_s);

    gemm128_wmma<<<gemmGrid, 256, 0, stream>>>(tb, nullptr, WtH, bt, qb, 1);  // t_out
    gemm128_wmma<<<gemmGrid, 256, 0, stream>>>(sb, nullptr, WsH, bsp, kb, 1); // s_out
    gemm128_wmma<<<gemmGrid, 256, 0, stream>>>(qb, kb, WoH, bo, out, 0);      // output
}
